// WitnessOverlap_30021821399043
// MI455X (gfx1250) — 8-bit matrix operand order repaired and checked by reading, compiled, not run on hardware
//
#include <hip/hip_runtime.h>


#define BATCH   2
#define STOK    1536
#define WITN    32
#define NVOCAB  4096
#define TPB     16
#define CHUNKB  (TPB * NVOCAB)
#define NTILES  (STOK / TPB)
#define NCHUNKS (BATCH * NTILES)

typedef int v8i __attribute__((ext_vector_type(8)));

__global__ __launch_bounds__(256)
void witness_hist_kernel(const int* __restrict__ idxq,
                         const int* __restrict__ idxk,
                         unsigned char* __restrict__ hq,
                         unsigned char* __restrict__ hk) {
    __shared__ __align__(16) unsigned int frag[CHUNKB / 4];
    __shared__ int toks[TPB * WITN];

    const int  tid = threadIdx.x;
    const int  blk = blockIdx.x;
    const bool isB = (blockIdx.y != 0);
    const int* src = isB ? idxk : idxq;
    unsigned char* dst = (isB ? hk : hq) + (size_t)blk * CHUNKB;

    for (int i = tid; i < CHUNKB / 4; i += 256) frag[i] = 0u;
    for (int i = tid; i < TPB * WITN; i += 256)
        toks[i] = src[blk * (TPB * WITN) + i];
    __syncthreads();

    for (int e = tid; e < TPB * WITN; e += 256) {
        const int r   = e >> 5;
        const int w   = e & 31;
        const int idx = toks[(r << 5) + w];
        int cnt = 0;
        bool first = true;
        #pragma unroll
        for (int j = 0; j < WITN; ++j) {
            const int m = (toks[(r << 5) + j] == idx);
            cnt += m;
            if (m && j < w) first = false;
        }
        if (first) {
            const int ks = idx >> 6;
            const int k6 = idx & 63;
            int lane, j;
            if (!isB) {
                lane = (r & 15) | (((k6 >> 4) & 1) << 4);
                j    = ((k6 >> 5) << 2) | ((k6 >> 2) & 3);
            } else {
                lane = (r & 15) | (((k6 >> 4) & 1) << 4);
                j    = ((k6 >> 5) << 2) | ((k6 >> 2) & 3);
            }
            ((unsigned char*)frag)[ks * 1024 + lane * 32 + j * 4 + (k6 & 3)]
                = (unsigned char)cnt;
        }
    }
    __syncthreads();

    const uint4* s4 = (const uint4*)frag;
    uint4*       d4 = (uint4*)dst;
    for (int i = tid; i < CHUNKB / 16; i += 256) d4[i] = s4[i];
}

__global__ __launch_bounds__(256)
void witness_gemm_kernel(const unsigned char* __restrict__ hq,
                         const unsigned char* __restrict__ hk,
                         float* __restrict__ out) {
    const int lane = threadIdx.x & 31;
    const int wid  = threadIdx.x >> 5;
    const int b    = blockIdx.z;
    const int m0   = blockIdx.y * 128 + (wid >> 2) * 64;
    const int n0   = blockIdx.x * 256 + (wid & 3) * 64;

    const unsigned char* aptr =
        hq + ((size_t)(b * NTILES + (m0 >> 4)) << 16) + lane * 32;
    const unsigned char* bptr =
        hk + ((size_t)(b * NTILES + (n0 >> 4)) << 16) + lane * 32;

    v8i acc[4][4];
    const v8i vzero = {0, 0, 0, 0, 0, 0, 0, 0};
    #pragma unroll
    for (int m = 0; m < 4; ++m)
        #pragma unroll
        for (int n = 0; n < 4; ++n) acc[m][n] = vzero;

    for (int ks = 0; ks < 64; ++ks) {
        v8i afr[4], bfr[4];
        #pragma unroll
        for (int m = 0; m < 4; ++m)
            afr[m] = *(const v8i*)(aptr + ((size_t)m << 16) + ks * 1024);
        #pragma unroll
        for (int n = 0; n < 4; ++n)
            bfr[n] = *(const v8i*)(bptr + ((size_t)n << 16) + ks * 1024);
        #pragma unroll
        for (int m = 0; m < 4; ++m)
            #pragma unroll
            for (int n = 0; n < 4; ++n)
                acc[m][n] = __builtin_amdgcn_wmma_i32_16x16x64_iu8(
                     false, afr[m],  false, bfr[n],
                    acc[m][n],  false,  false);
    }

    const int rhalf = (lane >> 4) << 3;
    const int col   = lane & 15;
    #pragma unroll
    for (int m = 0; m < 4; ++m)
        #pragma unroll
        for (int n = 0; n < 4; ++n)
            #pragma unroll
            for (int v = 0; v < 8; ++v) {
                const int row = m0 + m * 16 + v + rhalf;
                const int c   = n0 + n * 16 + col;
                out[((size_t)b * STOK + row) * (size_t)STOK + c]
                    = (float)acc[m][n][v];
            }
}

extern "C" void kernel_launch(void* const* d_in, const int* in_sizes, int n_in,
                              void* d_out, int out_size, void* d_ws, size_t ws_size,
                              hipStream_t stream) {
    const int* idxq = (const int*)d_in[0];
    const int* idxk = (const int*)d_in[1];
    float*     out  = (float*)d_out;

    unsigned char* hq = (unsigned char*)d_ws;
    unsigned char* hk = hq + (size_t)NCHUNKS * CHUNKB;

    witness_hist_kernel<<<dim3(NCHUNKS, 2, 1), 256, 0, stream>>>(idxq, idxk, hq, hk);

    witness_gemm_kernel<<<dim3(STOK / 256, STOK / 128, BATCH), 256, 0, stream>>>(hq, hk, out);
}
